// Attention_1657857376813
// MI455X (gfx1250) — compile-verified, matrix operand order checked by reading, not run on hardware
//
#include <hip/hip_runtime.h>

// Problem constants (from reference): B=8, N=1024, C=768, H=12, D=64
#define BB   8
#define NN   1024
#define CC   768
#define HH   12
#define DD   64
#define C3   2304
#define MM   (BB * NN)      // 8192 rows of x
#define SCALE 0.125f        // D^-0.5

typedef __bf16 bf16_t;
typedef __attribute__((ext_vector_type(16))) __bf16 v16bf;
typedef __attribute__((ext_vector_type(8)))  __bf16 v8bf;
typedef __attribute__((ext_vector_type(4)))  __bf16 v4bf;
typedef __attribute__((ext_vector_type(4)))  float  v4f;
typedef __attribute__((ext_vector_type(8)))  float  v8f;
typedef unsigned int u32;
typedef __attribute__((ext_vector_type(4))) u32 u32x4;
typedef __attribute__((ext_vector_type(4))) int  i32x4;
typedef __attribute__((ext_vector_type(8))) int  i32x8;

#ifndef __has_builtin
#define __has_builtin(x) 0
#endif
#if __has_builtin(__builtin_amdgcn_tensor_load_to_lds)
#define HAS_TDM 1
#else
#define HAS_TDM 0
#endif

__device__ __forceinline__ bf16_t to_bf16(float f) { return (bf16_t)f; }

__device__ __forceinline__ v8f wmma_bf16(v16bf a, v16bf b, v8f c) {
  return __builtin_amdgcn_wmma_f32_16x16x32_bf16(
      /*neg_a=*/false, a, /*neg_b=*/false, b,
      /*c_mod=*/(short)0, c, /*reuse_a=*/false, /*reuse_b=*/false);
}

// 16-bit A/B fragment (16x32 / 32x16): lane L holds row (L&15); group g = L>>4
// supplies K elements {g*8 .. g*8+7} and {g*8+16 .. g*8+23}.
__device__ __forceinline__ v16bf frag_bf16(const bf16_t* base, int stride,
                                           int row, int kk, int lane) {
  const bf16_t* p = base + (size_t)row * stride + kk + ((lane >> 4) << 3);
  v8bf lo = *reinterpret_cast<const v8bf*>(p);
  v8bf hi = *reinterpret_cast<const v8bf*>(p + 16);
  v16bf a;
#pragma unroll
  for (int i = 0; i < 8; ++i) { a[i] = lo[i]; a[8 + i] = hi[i]; }
  return a;
}

// Fragment from an LDS-resident 32-column tile (row stride 32 elements).
__device__ __forceinline__ v16bf frag_lds32(const bf16_t* tile, int row, int lane) {
  const bf16_t* p = tile + row * 32 + ((lane >> 4) << 3);
  v8bf lo = *reinterpret_cast<const v8bf*>(p);
  v8bf hi = *reinterpret_cast<const v8bf*>(p + 16);
  v16bf a;
#pragma unroll
  for (int i = 0; i < 8; ++i) { a[i] = lo[i]; a[8 + i] = hi[i]; }
  return a;
}

#if HAS_TDM
// TDM: DMA a 2-D tile (tile_d0 contiguous elems x tile_d1 rows, bf16) from
// global into LDS at byte offset lds_off.  D# bitfields per CDNA5 ISA ch.8.
__device__ __forceinline__ void tdm_load_tile_2d(u32 lds_off, const bf16_t* gptr,
                                                 u32 tile_d0, u32 tile_d1,
                                                 u32 tensor_d0, u32 stride0) {
  unsigned long long ga = (unsigned long long)(size_t)(const void*)gptr;
  u32 ga_lo = __builtin_amdgcn_readfirstlane((u32)ga);
  u32 ga_hi = __builtin_amdgcn_readfirstlane((u32)(ga >> 32));
  lds_off   = __builtin_amdgcn_readfirstlane(lds_off);
  u32x4 g0;
  g0[0] = 1u;                                  // count=1, user descriptor
  g0[1] = lds_off;                             // lds_addr (bytes)
  g0[2] = ga_lo;                               // global_addr[31:0]
  g0[3] = (ga_hi & 0x01ffffffu) | (2u << 30);  // global_addr[56:32] | type=2
  i32x8 g1;
  g1[0] = (int)(1u << 16);                     // wg_mask=0, data_size=1 (2B)
  g1[1] = (int)((tensor_d0 & 0xffffu) << 16);  // tensor_dim0[15:0]
  g1[2] = (int)((tensor_d0 >> 16) | (0xffffu << 16));      // td0 hi | td1 lo
  g1[3] = (int)((tile_d0 & 0xffffu) << 16);    // td1 hi=0 | tile_dim0
  g1[4] = (int)(tile_d1 & 0xffffu);            // tile_dim1 | tile_dim2=0
  g1[5] = (int)stride0;                        // tensor_dim0_stride[31:0]
  g1[6] = 0;                                   // stride0 hi | dim1_stride lo
  g1[7] = 0;
  i32x4 z4 = {};
#if defined(__clang_major__) && __clang_major__ >= 23
  i32x8 z8 = {};
  __builtin_amdgcn_tensor_load_to_lds(g0, g1, z4, z4, z8, 0);
#else
  __builtin_amdgcn_tensor_load_to_lds(g0, g1, z4, z4, 0);
#endif
}
#else
// Fallback: wave stages its 16x32 bf16 chunk via b128 loads/stores.
__device__ __forceinline__ void stage_chunk(bf16_t* dst, const bf16_t* src,
                                            int src_stride, int lane) {
#pragma unroll
  for (int t = 0; t < 2; ++t) {
    int seg = lane + 32 * t;             // 64 x 16B segments
    int row = seg >> 2, co = (seg & 3) * 8;
    *reinterpret_cast<v8bf*>(dst + row * 32 + co) =
        *reinterpret_cast<const v8bf*>(src + (size_t)row * src_stride + co);
  }
}
#endif

// ---------------------------------------------------------------------------
// Stage 0a: x f32 -> bf16, same layout, 4 elements/thread (vector cvt path)
// ---------------------------------------------------------------------------
__global__ void __launch_bounds__(256)
k_cvt_bf16(const float* __restrict__ src, bf16_t* __restrict__ dst, int n4) {
  int i = blockIdx.x * blockDim.x + threadIdx.x;
  if (i < n4) {
    v4f v = *reinterpret_cast<const v4f*>(src + 4 * (size_t)i);
    *reinterpret_cast<v4bf*>(dst + 4 * (size_t)i) = __builtin_convertvector(v, v4bf);
  }
}

// ---------------------------------------------------------------------------
// Stage 0b: transpose + convert weights: w[rows][cols] f32 -> wT[cols][rows]
// ---------------------------------------------------------------------------
__global__ void __launch_bounds__(256)
k_transpose_bf16(const float* __restrict__ w, bf16_t* __restrict__ wT,
                 int rows, int cols) {
  int i = blockIdx.x * blockDim.x + threadIdx.x;
  if (i < rows * cols) {
    int r = i / cols, c = i - r * cols;
    wT[(size_t)c * rows + r] = to_bf16(w[i]);
  }
}

// ===========================================================================
// Dense GEMM core: 128 threads = 4 waves; block computes 64 rows x 64 cols.
// B tile (64 x 32 bf16, 4 KB) staged in LDS by TDM (each wave DMAs its 16
// rows), double-buffered; DMA of step kk+32 overlaps WMMA of step kk.
// Wave w computes rows [m0+16w, m0+16w+16) against all 64 staged columns.
// ===========================================================================
#define GEMM_CORE(A_PTR, A_STRIDE, B_SRC_ROW_PTR)                              \
  __shared__ bf16_t Bt[2][64][32];                                             \
  const int lane = threadIdx.x & 31;                                           \
  const int wave = threadIdx.x >> 5;                                           \
  const int mloc = lane & 15, g = lane >> 4;                                   \
  const int m0 = blockIdx.x * 64 + wave * 16;                                  \
  const int n0 = blockIdx.y * 64;                                              \
  u32 lds_base = (u32)(size_t)(void*)&Bt[0][0][0];                             \
  (void)lds_base;                                                              \
  const bf16_t* bsrc = (B_SRC_ROW_PTR);                                        \
  v8f acc0 = {}, acc1 = {}, acc2 = {}, acc3 = {};                              \
  STAGE_B(0, 0);                                                               \
  int buf = 0;                                                                 \
  for (int kk = 0; kk < CC; kk += 32) {                                        \
    if (kk) __syncthreads();                                                   \
    if (kk + 32 < CC) { STAGE_B(buf ^ 1, kk + 32); }                           \
    WAIT_B(kk);                                                                \
    __syncthreads();                                                           \
    v16bf a  = frag_bf16((A_PTR), (A_STRIDE), m0 + mloc, kk, lane);            \
    v16bf b0 = frag_lds32(&Bt[buf][0][0],  mloc,      lane);                   \
    v16bf b1 = frag_lds32(&Bt[buf][0][0],  16 + mloc, lane);                   \
    v16bf b2 = frag_lds32(&Bt[buf][0][0],  32 + mloc, lane);                   \
    v16bf b3 = frag_lds32(&Bt[buf][0][0],  48 + mloc, lane);                   \
    acc0 = wmma_bf16(a, b0, acc0);                                             \
    acc1 = wmma_bf16(a, b1, acc1);                                             \
    acc2 = wmma_bf16(a, b2, acc2);                                             \
    acc3 = wmma_bf16(a, b3, acc3);                                             \
    buf ^= 1;                                                                  \
  }

#if HAS_TDM
#define STAGE_B(BUFI, KOFF)                                                    \
  tdm_load_tile_2d(lds_base + (u32)(BUFI) * 4096u + (u32)wave * 1024u,         \
                   bsrc + (KOFF), 32u, 16u, (u32)CC, (u32)CC)
#define WAIT_B(KK)                                                             \
  do {                                                                         \
    if ((KK) + 32 < CC) __builtin_amdgcn_s_wait_tensorcnt(1);                  \
    else                __builtin_amdgcn_s_wait_tensorcnt(0);                  \
  } while (0)
#else
#define STAGE_B(BUFI, KOFF)                                                    \
  stage_chunk(&Bt[(BUFI)][wave * 16][0], bsrc + (KOFF), CC, lane)
#define WAIT_B(KK) do { } while (0)
#endif

// ---------------------------------------------------------------------------
// Stage 1: QKV GEMM.  xb[8192,768] bf16 @ WqkvT -> q,k [B,H,N,D], v [B,H,D,N].
// ---------------------------------------------------------------------------
__global__ void __launch_bounds__(128)
k_qkv_gemm(const bf16_t* __restrict__ xb, const bf16_t* __restrict__ wT,
           bf16_t* __restrict__ q, bf16_t* __restrict__ kmat,
           bf16_t* __restrict__ vT) {
  GEMM_CORE(xb, CC, wT + (size_t)(blockIdx.y * 64 + (threadIdx.x >> 5) * 16) * CC)
  v8f accs[4] = {acc0, acc1, acc2, acc3};
#pragma unroll
  for (int t = 0; t < 4; ++t) {
    const int col = n0 + t * 16 + mloc;
    const int which = col / CC;          // uniform across wave per t
    const int jc = col - which * CC;
    const int h = jc >> 6, d = jc & 63;
#pragma unroll
    for (int r = 0; r < 8; ++r) {
      int m = m0 + r + 8 * g;
      int b_ = m >> 10, n = m & (NN - 1);
      bf16_t val = to_bf16(accs[t][r]);
      if (which == 0)      q   [(((b_ * HH + h) * NN + n) * DD) + d] = val;
      else if (which == 1) kmat[(((b_ * HH + h) * NN + n) * DD) + d] = val;
      else                 vT  [(((b_ * HH + h) * DD + d) * NN) + n] = val;
    }
  }
}

// ---------------------------------------------------------------------------
// Stage 3: output projection + bias, same TDM-staged GEMM core.
// ---------------------------------------------------------------------------
__global__ void __launch_bounds__(128)
k_proj_gemm(const bf16_t* __restrict__ ain, const bf16_t* __restrict__ wT,
            const float* __restrict__ bias, float* __restrict__ out) {
  GEMM_CORE(ain, CC, wT + (size_t)(blockIdx.y * 64 + (threadIdx.x >> 5) * 16) * CC)
  v8f accs[4] = {acc0, acc1, acc2, acc3};
#pragma unroll
  for (int t = 0; t < 4; ++t) {
    const int col = n0 + t * 16 + mloc;
    const float bb = bias[col];
#pragma unroll
    for (int r = 0; r < 8; ++r)
      out[(size_t)(m0 + r + 8 * g) * CC + col] = accs[t][r] + bb;
  }
}

// ---------------------------------------------------------------------------
// Stage 2: attention for one (b, h, 16-query stripe).  128 threads (4 waves).
// S (f32, 64 KB) + P (bf16, 32 KB) live in LDS -> the [B,H,N,N] matrix never
// touches HBM.  Softmax 1/sum folded into the P@V accumulator at writeout.
// ---------------------------------------------------------------------------
__global__ void __launch_bounds__(128)
k_attention(const bf16_t* __restrict__ q, const bf16_t* __restrict__ kmat,
            const bf16_t* __restrict__ vT, bf16_t* __restrict__ aout) {
  __shared__ float  S[16][NN];         // 64 KB raw scores
  __shared__ bf16_t P[16][NN];         // 32 KB probabilities (unnormalized)
  __shared__ float  invs[16];          // per-row 1/sum

  const int lane = threadIdx.x & 31;
  const int wave = threadIdx.x >> 5;   // 0..3
  const int q0 = blockIdx.x * 16;      // query row tile
  const int h  = blockIdx.y;
  const int b  = blockIdx.z;

  const bf16_t* qbase = q    + (size_t)((b * HH + h) * NN) * DD;
  const bf16_t* kbase = kmat + (size_t)((b * HH + h) * NN) * DD;
  const bf16_t* vbase = vT   + (size_t)((b * HH + h) * DD) * NN;
  const int g = lane >> 4, mloc = lane & 15;

  // ---- Phase 1: S = (q_tile @ K^T) * scale, 4 waves x 16 key tiles each ----
  v16bf a0 = frag_bf16(qbase, DD, q0 + mloc, 0, lane);
  v16bf a1 = frag_bf16(qbase, DD, q0 + mloc, 32, lane);
#pragma unroll 2
  for (int i = 0; i < 16; ++i) {
    int krow0 = (wave + 4 * i) * 16;
    v16bf b0 = frag_bf16(kbase, DD, krow0 + mloc, 0, lane);
    v16bf b1 = frag_bf16(kbase, DD, krow0 + mloc, 32, lane);
    v8f c = {};
    c = wmma_bf16(a0, b0, c);
    c = wmma_bf16(a1, b1, c);
#pragma unroll
    for (int r = 0; r < 8; ++r) S[r + 8 * g][krow0 + mloc] = c[r] * SCALE;
  }
  __syncthreads();

  // ---- Phase 2: row softmax; 8 lanes per row on contiguous 128-col chunks --
  {
    const int row = threadIdx.x >> 3;
    const int j = threadIdx.x & 7;
    const float* srow = &S[row][0];
    float mx = -3.0e38f;
    for (int c = j * 128; c < j * 128 + 128; c += 4) {
      v4f v = *reinterpret_cast<const v4f*>(srow + c);     // ds_load_b128
      mx = fmaxf(mx, fmaxf(fmaxf(v[0], v[1]), fmaxf(v[2], v[3])));
    }
    mx = fmaxf(mx, __shfl_xor(mx, 1, 8));
    mx = fmaxf(mx, __shfl_xor(mx, 2, 8));
    mx = fmaxf(mx, __shfl_xor(mx, 4, 8));
    float sum = 0.f;
    for (int c = j * 128; c < j * 128 + 128; c += 4) {
      v4f v = *reinterpret_cast<const v4f*>(srow + c);
      v4f e;
      e[0] = __expf(v[0] - mx); e[1] = __expf(v[1] - mx);
      e[2] = __expf(v[2] - mx); e[3] = __expf(v[3] - mx);
      sum += (e[0] + e[1]) + (e[2] + e[3]);
      *reinterpret_cast<v4bf*>(&P[row][c]) = __builtin_convertvector(e, v4bf);
    }
    sum += __shfl_xor(sum, 1, 8);
    sum += __shfl_xor(sum, 2, 8);
    sum += __shfl_xor(sum, 4, 8);
    if (j == 0) invs[row] = __frcp_rn(sum);
  }
  __syncthreads();

  // ---- Phase 3: O = P @ V; each wave owns one 16-wide d-tile --------------
  v8f c = {};
  for (int kk = 0; kk < NN; kk += 32) {
    const bf16_t* prow = &P[mloc][kk + (g << 3)];
    v8bf lo = *reinterpret_cast<const v8bf*>(prow);
    v8bf hi = *reinterpret_cast<const v8bf*>(prow + 16);
    v16bf a;
#pragma unroll
    for (int i = 0; i < 8; ++i) { a[i] = lo[i]; a[8 + i] = hi[i]; }
    v16bf bfr = frag_bf16(vbase, NN, wave * 16 + mloc, kk, lane);
    c = wmma_bf16(a, bfr, c);
  }
#pragma unroll
  for (int r = 0; r < 8; ++r) {
    int n = q0 + r + 8 * g;
    float o = c[r] * invs[r + 8 * g];   // fold softmax normalization here
    aout[((size_t)(b * NN + n) * CC) + h * DD + wave * 16 + mloc] = to_bf16(o);
  }
}

// ---------------------------------------------------------------------------
extern "C" void kernel_launch(void* const* d_in, const int* in_sizes, int n_in,
                              void* d_out, int out_size, void* d_ws, size_t ws_size,
                              hipStream_t stream) {
  const float* x      = (const float*)d_in[0];
  const float* w_qkv  = (const float*)d_in[1];
  const float* w_proj = (const float*)d_in[2];
  const float* b_proj = (const float*)d_in[3];
  float* out = (float*)d_out;

  // Workspace carve-up (all bf16): ~65 MB total
  bf16_t* xb     = (bf16_t*)d_ws;                      // [8192][768]
  bf16_t* wqkvT  = xb     + (size_t)MM * CC;           // [2304][768]
  bf16_t* wprojT = wqkvT  + (size_t)C3 * CC;           // [768][768]
  bf16_t* q      = wprojT + (size_t)CC * CC;           // [B,H,N,D]
  bf16_t* kmat   = q      + (size_t)BB * HH * NN * DD; // [B,H,N,D]
  bf16_t* vT     = kmat   + (size_t)BB * HH * NN * DD; // [B,H,D,N]
  bf16_t* aout   = vT     + (size_t)BB * HH * NN * DD; // [B,N,C]

  // Stage 0: input/weight conversion
  k_cvt_bf16<<<(MM * CC / 4 + 255) / 256, 256, 0, stream>>>(x, xb, MM * CC / 4);
  k_transpose_bf16<<<(CC * C3 + 255) / 256, 256, 0, stream>>>(w_qkv, wqkvT, CC, C3);
  k_transpose_bf16<<<(CC * CC + 255) / 256, 256, 0, stream>>>(w_proj, wprojT, CC, CC);

  // Stage 1: QKV GEMM (64-row x 64-col block tiles, TDM-staged B)
  k_qkv_gemm<<<dim3(MM / 64, C3 / 64), 128, 0, stream>>>(xb, wqkvT, q, kmat, vT);

  // Stage 2: attention (64 query tiles x 12 heads x 8 batches)
  k_attention<<<dim3(NN / 16, HH, BB), 128, 0, stream>>>(q, kmat, vT, aout);

  // Stage 3: projection + bias (TDM-staged B)
  k_proj_gemm<<<dim3(MM / 64, CC / 64), 128, 0, stream>>>(aout, wprojT, b_proj, out);
}
